// ConditionalRandomField_58677843198193
// MI455X (gfx1250) — matrix operand order repaired and checked by reading, not compiled, not run on hardware
//
#include <hip/hip_runtime.h>
#include <hip/hip_bf16.h>
#include <stdint.h>

#define CRF_BATCH 512
#define CRF_SEQ   1024
#define CRF_NT    50     // tags
#define CRF_NTP   64     // padded tags
#define CRF_BT    16     // batch tile per wave (= WMMA N)

typedef __attribute__((ext_vector_type(16))) _Float16 v16h;
typedef __attribute__((ext_vector_type(8)))  float    v8f;

union AFrag { v16h v; _Float16 e[16]; unsigned u[8]; };
union CFrag { v8f  v; float    e[8];  };
union H8    { uint4 q; _Float16 e[8]; };
union P2    { _Float16 e[2]; unsigned u; };

// Cross-half (lane <-> lane^16) exchange: v_permlanex16_b32 (VALU cross-row).
__device__ __forceinline__ int xchg16_i(int x) {
#if __has_builtin(__builtin_amdgcn_permlanex16)
  return __builtin_amdgcn_permlanex16(x, x, 0x76543210, 0xFEDCBA98, true, false);
#else
  return __shfl_xor(x, 16, 32);
#endif
}
__device__ __forceinline__ float    xchg16_f(float x)    { return __int_as_float(xchg16_i(__float_as_int(x))); }
__device__ __forceinline__ unsigned xchg16_u(unsigned x) { return (unsigned)xchg16_i((int)x); }

// Balanced-tree sum of a lane's 4x8 values: depth 5 instead of a 32-deep chain.
__device__ __forceinline__ float tree_sum32(const float v[4][8]) {
  float s[4];
  #pragma unroll
  for (int mb = 0; mb < 4; ++mb) {
    float a01 = v[mb][0] + v[mb][1];
    float a23 = v[mb][2] + v[mb][3];
    float a45 = v[mb][4] + v[mb][5];
    float a67 = v[mb][6] + v[mb][7];
    s[mb] = (a01 + a23) + (a45 + a67);
  }
  return (s[0] + s[1]) + (s[2] + s[3]);
}

// ---------------------------------------------------------------------------
// Pre-pass: emis[b][t][j] = exp(logits[b][t][j]) (f16), zero-padded to 64 tags.
// ---------------------------------------------------------------------------
__global__ __launch_bounds__(256) void crf_prepass_kernel(
    const float* __restrict__ logits, _Float16* __restrict__ emis)
{
  size_t idx = (size_t)blockIdx.x * 256 + threadIdx.x;
  if (idx >= (size_t)CRF_BATCH * CRF_SEQ * CRF_NTP) return;
  int j = (int)(idx & 63);
  size_t bt = idx >> 6;
  float v = (j < CRF_NT) ? __expf(logits[bt * CRF_NT + j]) : 0.0f;
  emis[idx] = (_Float16)v;
}

// ---------------------------------------------------------------------------
// Scaled forward algorithm, one wave per 16-batch tile.
//   D(j,b) = expT^T(j,i) x p^T(i,b)  -> 8x v_wmma_f32_16x16x32_f16
// Software-pipelined: emissions and mask byte for step t+1 are loaded while
// step t computes, so global-load latency is off the serial chain.
// ---------------------------------------------------------------------------
template <bool USE_PRE>
__global__ __launch_bounds__(32) void crf_forward_kernel(
    const float* __restrict__ logits, const uint8_t* __restrict__ mask,
    const float* __restrict__ trans,  const float* __restrict__ start_t,
    const float* __restrict__ end_t,  const _Float16* __restrict__ emis,
    float* __restrict__ den)
{
  const int lane = threadIdx.x;
  const int n0   = lane & 15;          // batch column within tile
  const int half = lane >> 4;          // 0: rows j%16<8 ; 1: j%16>=8
  const int b    = blockIdx.x * CRF_BT + n0;
  const size_t brow = (size_t)b * CRF_SEQ;

  // ---- resident A fragments: A[mb][kc] = expT^T tile (rows j, cols i) ----
  AFrag Aft[4][2];
  #pragma unroll
  for (int mb = 0; mb < 4; ++mb) {
    const int j = mb * 16 + n0;                  // output tag (A row)
    #pragma unroll
    for (int kc = 0; kc < 2; ++kc) {
      #pragma unroll
      for (int hh = 0; hh < 16; ++hh) {
        const int k = kc * 32 + 16 * half + hh;
        float v = (j < CRF_NT && k < CRF_NT) ? __expf(trans[k * CRF_NT + j]) : 0.0f;
        Aft[mb][kc].e[hh] = (_Float16)v;
      }
    }
  }

  // ---- pipeline prologue: emissions + mask for t = 1 ----
  H8   em[4];                 // USE_PRE path: f16 exp(logits) fragments
  float lraw[4][8];           // fallback path: raw logits
  uint8_t mk;
  if (USE_PRE) {
    const _Float16* erow = emis + (brow + 1) * CRF_NTP;
    #pragma unroll
    for (int mb = 0; mb < 4; ++mb)
      em[mb].q = *(const uint4*)(erow + mb * 16 + 8 * half);
  } else {
    const size_t base = (brow + 1) * CRF_NT;
    #pragma unroll
    for (int mb = 0; mb < 4; ++mb) {
      const int j0 = mb * 16 + 8 * half;
      #pragma unroll
      for (int r = 0; r < 8; ++r) {
        const int j = j0 + r;
        lraw[mb][r] = logits[base + ((j < CRF_NT) ? j : (CRF_NT - 1))];
      }
    }
  }
  mk = mask[brow + 1];

  // Lane's owned D entries: j = mb*16 + 8*half + r.
  // B fragment layout: Bf[kc].e[hh] = p[k = kc*32 + 16*half + hh][batch n0].
  AFrag Bf[2];
  float logZ;

  // ---- t = 0: alpha0 = start + logits[:,0]; normalize into Bf ----
  {
    float a[4][8];
    float mx = -1e30f;
    const size_t base0 = brow * CRF_NT;
    #pragma unroll
    for (int mb = 0; mb < 4; ++mb) {
      const int j0 = mb * 16 + 8 * half;
      #pragma unroll
      for (int r = 0; r < 8; ++r) {
        const int j = j0 + r;
        const int jc = (j < CRF_NT) ? j : (CRF_NT - 1);
        float v = (j < CRF_NT) ? (start_t[jc] + logits[base0 + jc]) : -1e30f;
        a[mb][r] = v;
        mx = fmaxf(mx, v);
      }
    }
    mx = fmaxf(mx, xchg16_f(mx));
    #pragma unroll
    for (int mb = 0; mb < 4; ++mb)
      #pragma unroll
      for (int r = 0; r < 8; ++r) a[mb][r] = __expf(a[mb][r] - mx);
    float rs = tree_sum32(a);
    rs += xchg16_f(rs);
    logZ = mx + __logf(rs);
    const float inv = __builtin_amdgcn_rcpf(rs);

    unsigned own[4][4];
    #pragma unroll
    for (int mb = 0; mb < 4; ++mb)
      #pragma unroll
      for (int rp = 0; rp < 4; ++rp) {
        P2 p; p.e[0] = (_Float16)(a[mb][2 * rp] * inv);
              p.e[1] = (_Float16)(a[mb][2 * rp + 1] * inv);
        own[mb][rp] = p.u;
      }
    unsigned recv0[4], recv1[4];
    #pragma unroll
    for (int i = 0; i < 4; ++i) {
      recv0[i] = xchg16_u(half ? own[0][i] : own[1][i]);
      recv1[i] = xchg16_u(half ? own[2][i] : own[3][i]);
    }
    #pragma unroll
    for (int i = 0; i < 4; ++i) {
      Bf[0].u[i]     = half ? recv0[i] : own[0][i];
      Bf[0].u[4 + i] = half ? own[1][i] : recv0[i];
      Bf[1].u[i]     = half ? recv1[i] : own[2][i];
      Bf[1].u[4 + i] = half ? own[3][i] : recv1[i];
    }
  }

  // ---- recurrence over t = 1 .. SEQ-1 (loads pipelined one step ahead) ----
  for (int t = 1; t < CRF_SEQ; ++t) {
    // Issue next step's loads first: a full step of compute hides them.
    H8   emn[4];
    float lrawn[4][8];
    uint8_t mkn = 0;
    if (t + 1 < CRF_SEQ) {
      if (USE_PRE) {
        const _Float16* ernext = emis + (brow + t + 1) * CRF_NTP;
        __builtin_prefetch(ernext + CRF_NTP, 0, 0);   // warm t+2
        #pragma unroll
        for (int mb = 0; mb < 4; ++mb)
          emn[mb].q = *(const uint4*)(ernext + mb * 16 + 8 * half);
      } else {
        const size_t basen = (brow + t + 1) * CRF_NT;
        __builtin_prefetch(&logits[basen + CRF_NT], 0, 0);
        #pragma unroll
        for (int mb = 0; mb < 4; ++mb) {
          const int j0 = mb * 16 + 8 * half;
          #pragma unroll
          for (int r = 0; r < 8; ++r) {
            const int j = j0 + r;
            lrawn[mb][r] = logits[basen + ((j < CRF_NT) ? j : (CRF_NT - 1))];
          }
        }
      }
      mkn = mask[brow + t + 1];
    }

    CFrag acc[4];
    #pragma unroll
    for (int mb = 0; mb < 4; ++mb) {
      v8f z = {};
      acc[mb].v = z;
      acc[mb].v = __builtin_amdgcn_wmma_f32_16x16x32_f16(
          false, Aft[mb][0].v, false, Bf[0].v, (short)0, acc[mb].v, false, false);
      acc[mb].v = __builtin_amdgcn_wmma_f32_16x16x32_f16(
          false, Aft[mb][1].v, false, Bf[1].v, (short)0, acc[mb].v, false, false);
    }

    // emission * q, then tree rowsum over this lane's 32 tag entries
    float qe[4][8];
    if (USE_PRE) {
      #pragma unroll
      for (int mb = 0; mb < 4; ++mb)
        #pragma unroll
        for (int r = 0; r < 8; ++r)
          qe[mb][r] = acc[mb].e[r] * (float)em[mb].e[r];
    } else {
      #pragma unroll
      for (int mb = 0; mb < 4; ++mb) {
        const int j0 = mb * 16 + 8 * half;
        #pragma unroll
        for (int r = 0; r < 8; ++r) {
          const int j = j0 + r;
          float e = (j < CRF_NT) ? __expf(lraw[mb][r]) : 0.f;
          qe[mb][r] = acc[mb].e[r] * e;
        }
      }
    }
    float rs = tree_sum32(qe);
    rs += xchg16_f(rs);

    const bool upd = mk != 0;              // same for both half-lanes of a batch
    if (upd) logZ += __logf(rs);
    const float inv = __builtin_amdgcn_rcpf(rs);

    // pack new p, exchange halves, assemble next B fragments (masked commit)
    unsigned own[4][4];
    #pragma unroll
    for (int mb = 0; mb < 4; ++mb)
      #pragma unroll
      for (int rp = 0; rp < 4; ++rp) {
        P2 p; p.e[0] = (_Float16)(qe[mb][2 * rp] * inv);
              p.e[1] = (_Float16)(qe[mb][2 * rp + 1] * inv);
        own[mb][rp] = p.u;
      }
    unsigned recv0[4], recv1[4];
    #pragma unroll
    for (int i = 0; i < 4; ++i) {
      recv0[i] = xchg16_u(half ? own[0][i] : own[1][i]);
      recv1[i] = xchg16_u(half ? own[2][i] : own[3][i]);
    }
    if (upd) {
      #pragma unroll
      for (int i = 0; i < 4; ++i) {
        Bf[0].u[i]     = half ? recv0[i] : own[0][i];
        Bf[0].u[4 + i] = half ? own[1][i] : recv0[i];
        Bf[1].u[i]     = half ? recv1[i] : own[2][i];
        Bf[1].u[4 + i] = half ? own[3][i] : recv1[i];
      }
    }

    // rotate pipeline buffers
    if (USE_PRE) {
      #pragma unroll
      for (int mb = 0; mb < 4; ++mb) em[mb] = emn[mb];
    } else {
      #pragma unroll
      for (int mb = 0; mb < 4; ++mb)
        #pragma unroll
        for (int r = 0; r < 8; ++r) lraw[mb][r] = lrawn[mb][r];
    }
    mk = mkn;
  }

  // ---- finalize: den = logZ + log(sum_k p[k] * exp(end[k])) ----
  {
    float fin[4][8];
    #pragma unroll
    for (int kc = 0; kc < 2; ++kc)
      #pragma unroll
      for (int hh = 0; hh < 16; ++hh) {
        const int k = kc * 32 + 16 * half + hh;
        float v = (k < CRF_NT) ? (float)Bf[kc].e[hh] * __expf(end_t[k]) : 0.f;
        fin[kc * 2 + (hh >> 3)][hh & 7] = v;
      }
    float rs = tree_sum32(fin);
    rs += xchg16_f(rs);
    if (half == 0) den[b] = logZ + __logf(rs);
  }
}

// ---------------------------------------------------------------------------
// Numerator: masked path score.  One 256-thread block per batch row.
// ---------------------------------------------------------------------------
__global__ __launch_bounds__(256) void crf_numerator_kernel(
    const float* __restrict__ logits, const int* __restrict__ tags,
    const uint8_t* __restrict__ mask, const float* __restrict__ trans,
    const float* __restrict__ start_t, const float* __restrict__ end_t,
    float* __restrict__ num)
{
  __shared__ float red[256];
  __shared__ int   cnt[256];
  const int b = blockIdx.x, tid = threadIdx.x;
  const size_t bs = (size_t)b * CRF_SEQ;

  float part = 0.f;
  int mc = 0;
  for (int t = tid; t < CRF_SEQ; t += 256) {
    const int mt = mask[bs + t] ? 1 : 0;
    mc += mt;
    if (t < CRF_SEQ - 1) {
      const int tg0 = tags[bs + t];
      const int tg1 = tags[bs + t + 1];
      const float f1 = mask[bs + t + 1] ? 1.f : 0.f;
      part += trans[tg0 * CRF_NT + tg1] * f1
            + logits[(bs + t) * CRF_NT + tg0] * (float)mt;
    }
  }
  red[tid] = part; cnt[tid] = mc;
  __syncthreads();
  for (int s = 128; s > 0; s >>= 1) {
    if (tid < s) { red[tid] += red[tid + s]; cnt[tid] += cnt[tid + s]; }
    __syncthreads();
  }
  if (tid == 0) {
    float score = red[0] + start_t[tags[bs]];
    const int last = cnt[0] - 1;                 // sum(mask) - 1
    const int lt = tags[bs + last];
    const float fl = mask[bs + CRF_SEQ - 1] ? 1.f : 0.f;
    score += end_t[lt] + logits[(bs + CRF_SEQ - 1) * CRF_NT + lt] * fl;
    num[b] = score;
  }
}

// ---------------------------------------------------------------------------
// Final scalar: sum_b (num[b] - den[b])
// ---------------------------------------------------------------------------
__global__ __launch_bounds__(512) void crf_reduce_kernel(
    const float* __restrict__ num, const float* __restrict__ den,
    float* __restrict__ out)
{
  __shared__ float s[512];
  const int i = threadIdx.x;
  s[i] = num[i] - den[i];
  __syncthreads();
  for (int st = 256; st > 0; st >>= 1) {
    if (i < st) s[i] += s[i + st];
    __syncthreads();
  }
  if (i == 0) out[0] = s[0];
}

extern "C" void kernel_launch(void* const* d_in, const int* in_sizes, int n_in,
                              void* d_out, int out_size, void* d_ws, size_t ws_size,
                              hipStream_t stream) {
  const float*   logits  = (const float*)d_in[0];
  const int*     tags    = (const int*)d_in[1];
  const uint8_t* mask    = (const uint8_t*)d_in[2];   // jnp bool -> 1 byte
  const float*   trans   = (const float*)d_in[3];
  const float*   start_t = (const float*)d_in[4];
  const float*   end_t   = (const float*)d_in[5];
  float* out = (float*)d_out;

  float* num = (float*)d_ws;                        // [512]
  float* den = num + CRF_BATCH;                     // [512]
  _Float16* emis = (_Float16*)((char*)d_ws + 4096); // padded exp(logits), f16

  const size_t emisBytes = (size_t)CRF_BATCH * CRF_SEQ * CRF_NTP * sizeof(_Float16);
  const bool use_pre = ws_size >= 4096 + emisBytes; // deterministic per config

  crf_numerator_kernel<<<CRF_BATCH, 256, 0, stream>>>(
      logits, tags, mask, trans, start_t, end_t, num);

  if (use_pre) {
    const size_t total = (size_t)CRF_BATCH * CRF_SEQ * CRF_NTP;
    crf_prepass_kernel<<<(unsigned)((total + 255) / 256), 256, 0, stream>>>(logits, emis);
    crf_forward_kernel<true><<<CRF_BATCH / CRF_BT, 32, 0, stream>>>(
        logits, mask, trans, start_t, end_t, emis, den);
  } else {
    crf_forward_kernel<false><<<CRF_BATCH / CRF_BT, 32, 0, stream>>>(
        logits, mask, trans, start_t, end_t, nullptr, den);
  }

  crf_reduce_kernel<<<1, CRF_BATCH, 0, stream>>>(num, den, out);
}
